// EMACode_17428977287705
// MI455X (gfx1250) — compile-verified, matrix operand order checked by reading, not run on hardware
//
#include <hip/hip_runtime.h>
#include <hip/hip_bf16.h>

typedef __attribute__((ext_vector_type(16))) __bf16 v16bf;
typedef __attribute__((ext_vector_type(8)))  float  v8f;

#define LAMDA 0.999f
#define DDIM 64
#define KC   128      // codes staged in LDS per chunk (16 KB bf16)
#define KMAX 2048     // codebook size upper bound for LDS c2 cache

// ---------------------------------------------------------------------------
// prep: embed f32 -> bf16, c2[k] = sum(embed[k]^2), zero hits & z_sum
// grid = K blocks x 64 threads
__global__ void prep_kernel(const float* __restrict__ embed,
                            __bf16* __restrict__ embed_bf,
                            float* __restrict__ c2,
                            float* __restrict__ hits,
                            float* __restrict__ zsum) {
  const int k = blockIdx.x;
  const int d = threadIdx.x;           // 0..63
  const float e = embed[(size_t)k * DDIM + d];
  embed_bf[(size_t)k * DDIM + d] = (__bf16)e;
  zsum[(size_t)k * DDIM + d] = 0.0f;
  float s = e * e;
  #pragma unroll
  for (int off = 16; off > 0; off >>= 1) s += __shfl_xor(s, off, 32);
  __shared__ float part[2];
  if ((threadIdx.x & 31) == 0) part[threadIdx.x >> 5] = s;
  __syncthreads();
  if (threadIdx.x == 0) {
    c2[k] = part[0] + part[1];
    hits[k] = 0.0f;
  }
}

// ---------------------------------------------------------------------------
// async staging: one chunk (KC*DDIM bf16 = 16 KB) via 4 async b128 per thread.
// INST_OFFSET is added to BOTH the LDS and global addresses (ISA 08 §4.4),
// so one LDS-addr VGPR + one global-offset VGPR cover all four issues.
__device__ __forceinline__ void issue_chunk_async(const __bf16* base,
                                                  int chunk,
                                                  unsigned lds_byte_base) {
  const unsigned go = (unsigned)(chunk * (KC * DDIM * 2)) + threadIdx.x * 16u;
  const unsigned lo = lds_byte_base + threadIdx.x * 16u;
  asm volatile("global_load_async_to_lds_b128 %0, %1, %2 offset:0"
               :: "v"(lo), "v"(go), "s"(base) : "memory");
  asm volatile("global_load_async_to_lds_b128 %0, %1, %2 offset:4096"
               :: "v"(lo), "v"(go), "s"(base) : "memory");
  asm volatile("global_load_async_to_lds_b128 %0, %1, %2 offset:8192"
               :: "v"(lo), "v"(go), "s"(base) : "memory");
  asm volatile("global_load_async_to_lds_b128 %0, %1, %2 offset:12288"
               :: "v"(lo), "v"(go), "s"(base) : "memory");
}

__device__ __forceinline__ void wait_async0() {
#if __has_builtin(__builtin_amdgcn_s_wait_asynccnt)
  __builtin_amdgcn_s_wait_asynccnt(0);
#else
  asm volatile("s_wait_asynccnt 0" ::: "memory");
#endif
}

// ---------------------------------------------------------------------------
// vq: fused GEMM (x @ embed^T) + row-wise argmin of (c2 - 2*score).
// We maximize q = score - 0.5*c2 (same argmin).
// Block = 256 threads = 8 waves; wave w owns rows [blk*128 + w*16, +16).
// Double-buffered LDS chunks filled with GLOBAL_LOAD_ASYNC_TO_LDS_B128.
__global__ void __launch_bounds__(256)
vq_kernel(const float* __restrict__ x,          // [N,64]
          const __bf16* __restrict__ embed_bf,  // [K,64]
          const float* __restrict__ c2,         // [K]
          int*   __restrict__ idx_out,          // [N] (ws)
          float* __restrict__ idx_f_out,        // [N] (output, float)
          int K) {
  __shared__ __align__(16) __bf16 sB[2][KC * DDIM];  // 2 x 16 KB
  __shared__ float sH[KMAX];                         // 0.5*c2, whole codebook

  const int lane = threadIdx.x & 31;
  const int wave = threadIdx.x >> 5;
  const int rowBase = blockIdx.x * 128 + wave * 16;
  const int half = lane >> 4;     // 0: lanes 0-15, 1: lanes 16-31
  const int l16  = lane & 15;

  // 16-bit A/B fragment K-column groups per ISA 7.12.2:
  // lanes 0-15 hold K {0..7, 16..23}; lanes 16-31 hold K {8..15, 24..31}
  const int cA = half ? 8  : 0;
  const int cB = half ? 24 : 16;

  // generic->LDS: low 32 bits of a __shared__ address are the LDS byte offset
  const unsigned sB0 = (unsigned)(size_t)(&sB[0][0]);
  const unsigned sB1 = (unsigned)(size_t)(&sB[1][0]);

  // preload all of 0.5*c2 into LDS (once), and kick off chunk 0 async fill
  issue_chunk_async(embed_bf, 0, sB0);
  for (int i = threadIdx.x; i < K; i += 256) sH[i] = 0.5f * c2[i];

  // ---- load A fragments (this wave's 16 rows, K-dim = 64 -> a0,a1) ----
  v16bf a0, a1;
  {
    const float* xr = x + (size_t)(rowBase + l16) * DDIM;
    #pragma unroll
    for (int i = 0; i < 8; ++i) {
      a0[i]     = (__bf16)xr[cA + i];
      a0[8 + i] = (__bf16)xr[cB + i];
      a1[i]     = (__bf16)xr[32 + cA + i];
      a1[8 + i] = (__bf16)xr[32 + cB + i];
    }
  }

  float best[8];
  int   bidx[8];
  #pragma unroll
  for (int r = 0; r < 8; ++r) { best[r] = -3.4e38f; bidx[r] = 0; }

  wait_async0();
  __syncthreads();

  const int nchunks = K / KC;
  for (int c = 0; c < nchunks; ++c) {
    // overlap: start filling the other buffer while we compute on this one
    if (c + 1 < nchunks)
      issue_chunk_async(embed_bf, c + 1, (c & 1) ? sB0 : sB1);

    const __bf16* sbuf = sB[c & 1];
    const int k0 = c * KC;

    #pragma unroll 2
    for (int t = 0; t < KC; t += 16) {
      const int codeN = t + l16;               // this lane's output column
      const __bf16* br = sbuf + codeN * DDIM;  // B column n = embed row codeN
      v16bf b0, b1;
      ((uint4*)&b0)[0] = *(const uint4*)(br + cA);
      ((uint4*)&b0)[1] = *(const uint4*)(br + cB);
      ((uint4*)&b1)[0] = *(const uint4*)(br + 32 + cA);
      ((uint4*)&b1)[1] = *(const uint4*)(br + 32 + cB);

      v8f acc = {};
      acc = __builtin_amdgcn_wmma_f32_16x16x32_bf16(false, a0, false, b0,
                                                    (short)0, acc, false, false);
      acc = __builtin_amdgcn_wmma_f32_16x16x32_bf16(false, a1, false, b1,
                                                    (short)0, acc, false, false);

      const float h = sH[k0 + codeN];
      const int code = k0 + codeN;
      #pragma unroll
      for (int r = 0; r < 8; ++r) {
        const float q = acc[r] - h;          // maximize q == minimize dist
        const bool better = q > best[r];     // strict: earlier code wins ties
        best[r] = better ? q : best[r];
        bidx[r] = better ? code : bidx[r];
      }
    }

    if (c + 1 < nchunks) {
      wait_async0();       // my async LDS fills for chunk c+1 have landed
      __syncthreads();     // everyone's have; buffer switch is safe
    }
  }

  // ---- cross-lane argmax over the 16-lane N group (C layout: VGPR r ->
  // row M=r for lanes 0-15, M=8+r for lanes 16-31) ----
  #pragma unroll
  for (int r = 0; r < 8; ++r) {
    float v = best[r]; int ix = bidx[r];
    #pragma unroll
    for (int off = 1; off < 16; off <<= 1) {
      const float ov = __shfl_xor(v, off, 32);
      const int   oi = __shfl_xor(ix, off, 32);
      if (ov > v || (ov == v && oi < ix)) { v = ov; ix = oi; }
    }
    best[r] = v; bidx[r] = ix;
  }
  if (l16 == 0) {
    const int rbase = rowBase + half * 8;
    #pragma unroll
    for (int r = 0; r < 8; ++r) {
      idx_out[rbase + r]   = bidx[r];
      idx_f_out[rbase + r] = (float)bidx[r];
    }
  }
}

// ---------------------------------------------------------------------------
// scatter: z_q gather + masked segment sums via f32 atomics
// thread t -> (row = t/64, d = t%64)
__global__ void scatter_kernel(const float* __restrict__ x,
                               const float* __restrict__ embed,
                               const int* __restrict__ idx,
                               const int* __restrict__ mask,
                               float* __restrict__ zq,
                               float* __restrict__ hits,
                               float* __restrict__ zsum, int N) {
  const int t = blockIdx.x * 256 + threadIdx.x;
  const int row = t >> 6;
  const int d = t & 63;
  if (row >= N) return;
  const int k = idx[row];
  zq[(size_t)row * DDIM + d] = embed[(size_t)k * DDIM + d];
  if (mask[row] != 0) {
    atomicAdd(&zsum[(size_t)k * DDIM + d], x[(size_t)row * DDIM + d]);
    if (d == 0) atomicAdd(&hits[k], 1.0f);
  }
}

// ---------------------------------------------------------------------------
// finalize: EMA update (matches reference exactly)
__global__ void finalize_kernel(const float* __restrict__ embed,
                                const float* __restrict__ code_count,
                                const float* __restrict__ hits,
                                const float* __restrict__ zsum,
                                float* __restrict__ new_code,
                                float* __restrict__ new_count) {
  const int k = blockIdx.x;
  const int d = threadIdx.x;
  const float cnt = LAMDA * code_count[k] + (1.0f - LAMDA) * hits[k];
  const float e = embed[(size_t)k * DDIM + d];
  float out;
  if (cnt > 0.0f) {
    out = e * LAMDA + (1.0f - LAMDA) * zsum[(size_t)k * DDIM + d] / cnt;
  } else {
    out = e;
  }
  new_code[(size_t)k * DDIM + d] = out;
  if (d == 0) new_count[k] = cnt;
}

// ---------------------------------------------------------------------------
extern "C" void kernel_launch(void* const* d_in, const int* in_sizes, int n_in,
                              void* d_out, int out_size, void* d_ws, size_t ws_size,
                              hipStream_t stream) {
  const float* x          = (const float*)d_in[0];   // [B,T,D]
  const float* embed      = (const float*)d_in[1];   // [K,D]
  const float* code_count = (const float*)d_in[2];   // [K]
  const int*   mask       = (const int*)d_in[3];     // [B,T]

  const int K = in_sizes[2];   // 2048
  const int N = in_sizes[3];   // B*T = 131072

  // output layout (all f32): z_q | indices | new_code | new_count
  float* out       = (float*)d_out;
  float* zq        = out;
  float* idx_f     = zq + (size_t)N * DDIM;
  float* new_code  = idx_f + N;
  float* new_count = new_code + (size_t)K * DDIM;

  // workspace layout
  char* ws = (char*)d_ws;
  __bf16* embed_bf = (__bf16*)ws;  ws += (size_t)K * DDIM * 2;
  float*  c2       = (float*)ws;   ws += (size_t)K * 4;
  float*  hits     = (float*)ws;   ws += (size_t)K * 4;
  float*  zsum     = (float*)ws;   ws += (size_t)K * DDIM * 4;
  int*    idx      = (int*)ws;     ws += (size_t)N * 4;

  prep_kernel<<<K, 64, 0, stream>>>(embed, embed_bf, c2, hits, zsum);
  vq_kernel<<<N / 128, 256, 0, stream>>>(x, embed_bf, c2, idx, idx_f, K);
  scatter_kernel<<<(N * DDIM) / 256, 256, 0, stream>>>(x, embed, idx, mask,
                                                       zq, hits, zsum, N);
  finalize_kernel<<<K, 64, 0, stream>>>(embed, code_count, hits, zsum,
                                        new_code, new_count);
}
